// AttentiveBP_19567871001287
// MI455X (gfx1250) — matrix operand order repaired and checked by reading, not compiled, not run on hardware
//
#include <hip/hip_runtime.h>

typedef __attribute__((ext_vector_type(16))) _Float16 v16h;
typedef __attribute__((ext_vector_type(8)))  float    v8f;

// ---- problem constants (from reference setup_inputs) ----
#define K_NX 100000
#define K_A  50000
#define K_B  50000
#define K_N  200000          // total graph nodes (Nx + A + B)
#define K_IC 16
#define K_PD 3
#define K_GH 13              // IC - PD
#define K_E  2000000
#define K_ET (K_E + K_N)     // edges + self loops
#define K_M  1024
#define K_G  1024
#define K_L  4
#define K_H  4               // GAT heads
#define K_OC 32

// ---------------- device helpers ----------------
__device__ __forceinline__ float lrelu(float v, float s) { return v > 0.f ? v : s * v; }
__device__ __forceinline__ float sigm(float v) { return 1.f / (1.f + __expf(-v)); }

// monotone float -> uint key for atomicMax on floats
__device__ __forceinline__ unsigned fkey(float f) {
  unsigned u = __float_as_uint(f);
  return (u & 0x80000000u) ? ~u : (u | 0x80000000u);
}
__device__ __forceinline__ float funkey(unsigned k) {
  unsigned u = (k & 0x80000000u) ? (k ^ 0x80000000u) : ~k;
  return __uint_as_float(u);
}

__device__ __forceinline__ void edge_sd(const int* ei, int e, int& s, int& d) {
  if (e < K_E) { s = ei[e]; d = ei[K_E + e]; }
  else         { s = e - K_E; d = s; }
}

// ---------------- GRU cell ----------------
__global__ void gru_kernel(const float* __restrict__ x, const float* __restrict__ h,
                           const float* __restrict__ Wih, const float* __restrict__ Whh,
                           const float* __restrict__ bih, const float* __restrict__ bhh,
                           float* __restrict__ hout, int n) {
  int i = blockIdx.x * blockDim.x + threadIdx.x;
  if (i >= n) return;
  float xv = x[i];
  float hv[K_GH];
#pragma unroll
  for (int j = 0; j < K_GH; ++j) hv[j] = h[i * K_GH + j];
#pragma unroll
  for (int j = 0; j < K_GH; ++j) {
    float gi_r = xv * Wih[j]            + bih[j];
    float gi_z = xv * Wih[K_GH + j]     + bih[K_GH + j];
    float gi_n = xv * Wih[2 * K_GH + j] + bih[2 * K_GH + j];
    float gh_r = bhh[j], gh_z = bhh[K_GH + j], gh_n = bhh[2 * K_GH + j];
#pragma unroll
    for (int k = 0; k < K_GH; ++k) {
      gh_r += hv[k] * Whh[j * K_GH + k];
      gh_z += hv[k] * Whh[(K_GH + j) * K_GH + k];
      gh_n += hv[k] * Whh[(2 * K_GH + j) * K_GH + k];
    }
    float r  = sigm(gi_r + gh_r);
    float z  = sigm(gi_z + gh_z);
    float nn = tanhf(gi_n + r * gh_n);
    hout[i * K_GH + j] = (1.f - z) * nn + z * hv[j];
  }
}

// ---------------- assemble node features (stride 32, cols 16..31 zero) ----------------
__global__ void assemble_feat(const float* __restrict__ x, const float* __restrict__ prefix,
                              const float* __restrict__ h1, const float* __restrict__ h2,
                              float* __restrict__ feat) {
  int n = blockIdx.x * blockDim.x + threadIdx.x;
  if (n >= K_N) return;
  float f[K_IC];
  if (n < K_NX) {
#pragma unroll
    for (int c = 0; c < K_IC; ++c) f[c] = x[(size_t)n * K_IC + c];
  } else {
    int i = n - K_NX;
#pragma unroll
    for (int c = 0; c < K_PD; ++c) f[c] = prefix[(size_t)i * K_PD + c];
    const float* hh = (i < K_A) ? &h1[(size_t)i * K_GH] : &h2[(size_t)(i - K_A) * K_GH];
#pragma unroll
    for (int j = 0; j < K_GH; ++j) f[K_PD + j] = hh[j];
  }
  float* fr = feat + (size_t)n * 32;
#pragma unroll
  for (int c = 0; c < K_IC; ++c) fr[c] = f[c];
#pragma unroll
  for (int c = K_IC; c < 32; ++c) fr[c] = 0.f;
}

// ---------------- pack W[K x Nc] (f32) into per-lane B-fragment order (f16) ----------------
// Wh[((ct*32 + lane)*16 + i] = W[k][col], k = (i>>3)*16 + (lane>>4)*8 + (i&7), col = ct*16 + (lane&15)
__global__ void pack_w(const float* __restrict__ W, int K, int Nc, _Float16* __restrict__ Wh) {
  int t = blockIdx.x * blockDim.x + threadIdx.x;
  if (t >= Nc * 32) return;
  int i    = t & 15;
  int lane = (t >> 4) & 31;
  int ct   = t >> 9;
  int k    = (i >> 3) * 16 + (lane >> 4) * 8 + (i & 7);
  int col  = ct * 16 + (lane & 15);
  int kk   = (k < K) ? k : 0;
  float v  = W[(size_t)kk * Nc + col];
  Wh[t] = (k < K) ? (_Float16)v : (_Float16)0.f;
}

// ---------------- WMMA feature transform: H[N x Nc] = X[N x 32] @ W[32 x Nc] ----------------
// One wave per 16x16 output tile; A fed by 4x b128 loads, B by one 32B packed load.
__global__ __launch_bounds__(32)
void wmma_mm(const float* __restrict__ X, const _Float16* __restrict__ Wh,
             int Nc, float* __restrict__ H) {
  int rowbase = blockIdx.x * 16;
  int ct   = blockIdx.y;
  int lane = threadIdx.x;            // 0..31
  int lo16 = lane & 15;
  int kgrp = lane >> 4;              // 0 or 1

  // A: 16x32 f16. lanes 0-15: K {0..7, 16..23}; lanes 16-31: K {8..15, 24..31}
  const float4* xr = (const float4*)(X + (size_t)(rowbase + lo16) * 32);
  float4 p0 = xr[kgrp * 2 + 0];      // k = kgrp*8 + 0..3
  float4 p1 = xr[kgrp * 2 + 1];      // k = kgrp*8 + 4..7
  float4 p2 = xr[4 + kgrp * 2 + 0];  // k = 16 + kgrp*8 + 0..3
  float4 p3 = xr[4 + kgrp * 2 + 1];  // k = 16 + kgrp*8 + 4..7
  v16h a;
  a[0]  = (_Float16)p0.x; a[1]  = (_Float16)p0.y; a[2]  = (_Float16)p0.z; a[3]  = (_Float16)p0.w;
  a[4]  = (_Float16)p1.x; a[5]  = (_Float16)p1.y; a[6]  = (_Float16)p1.z; a[7]  = (_Float16)p1.w;
  a[8]  = (_Float16)p2.x; a[9]  = (_Float16)p2.y; a[10] = (_Float16)p2.z; a[11] = (_Float16)p2.w;
  a[12] = (_Float16)p3.x; a[13] = (_Float16)p3.y; a[14] = (_Float16)p3.z; a[15] = (_Float16)p3.w;

  // B: pre-packed fragment, one 32-byte vector load per lane
  v16h b = *(const v16h*)(Wh + ((size_t)ct * 32 + lane) * 16);

  v8f c = {};
  c = __builtin_amdgcn_wmma_f32_16x16x32_f16(false, a, false, b, (short)0, c, false, false);

  // D: f32 16x16. VGPR r, lanes 0-15 -> M = r, lanes 16-31 -> M = r+8; col = lane&15
  int bcol = ct * 16 + lo16;
#pragma unroll
  for (int r = 0; r < 8; ++r) {
    H[(size_t)(rowbase + kgrp * 8 + r) * Nc + bcol] = c[r];
  }
}

// ---------------- per-node attention coefficients ----------------
__global__ void attn_coef(const float* __restrict__ Hf, int Cc,
                          const float* __restrict__ asrc, const float* __restrict__ adst,
                          float* __restrict__ as_, float* __restrict__ ad_) {
  int t = blockIdx.x * blockDim.x + threadIdx.x;   // n*H + hd
  if (t >= K_N * K_H) return;
  int n = t / K_H, hd = t % K_H;
  const float* hp = Hf + (size_t)n * (K_H * Cc) + hd * Cc;
  float s1 = 0.f, s2 = 0.f;
  for (int c = 0; c < Cc; ++c) { s1 += hp[c] * asrc[hd * Cc + c]; s2 += hp[c] * adst[hd * Cc + c]; }
  as_[t] = s1; ad_[t] = s2;
}

__global__ void fill_kernel(float* __restrict__ p, float v, int n) {
  int t = blockIdx.x * blockDim.x + threadIdx.x;
  if (t < n) p[t] = v;
}

// ---------------- edge passes ----------------
__global__ void edge_max(const int* __restrict__ ei, const float* __restrict__ as_,
                         const float* __restrict__ ad_, unsigned* __restrict__ m) {
  int e = blockIdx.x * blockDim.x + threadIdx.x;
  if (e >= K_ET) return;
  int s, d; edge_sd(ei, e, s, d);
#pragma unroll
  for (int hd = 0; hd < K_H; ++hd) {
    float v = lrelu(as_[s * K_H + hd] + ad_[d * K_H + hd], 0.2f);
    atomicMax(&m[d * K_H + hd], fkey(v));
  }
}

__global__ void edge_expsum(const int* __restrict__ ei, const float* __restrict__ as_,
                            const float* __restrict__ ad_, const unsigned* __restrict__ m,
                            float* __restrict__ den) {
  int e = blockIdx.x * blockDim.x + threadIdx.x;
  if (e >= K_ET) return;
  int s, d; edge_sd(ei, e, s, d);
#pragma unroll
  for (int hd = 0; hd < K_H; ++hd) {
    float v = lrelu(as_[s * K_H + hd] + ad_[d * K_H + hd], 0.2f);
    atomicAdd(&den[d * K_H + hd], __expf(v - funkey(m[d * K_H + hd])));
  }
}

__global__ void edge_aggr(const int* __restrict__ ei, const float* __restrict__ as_,
                          const float* __restrict__ ad_, const unsigned* __restrict__ m,
                          const float* __restrict__ den, const float* __restrict__ Hf,
                          int Cc, float* __restrict__ agg, int concat, float scale) {
  int e = blockIdx.x * blockDim.x + threadIdx.x;
  if (e >= K_ET) return;
  int s, d; edge_sd(ei, e, s, d);
  // random 128B-512B gather of the source feature row dominates; prefetch it
  __builtin_prefetch(Hf + (size_t)s * (K_H * Cc), 0, 3);
#pragma unroll
  for (int hd = 0; hd < K_H; ++hd) {
    float v = lrelu(as_[s * K_H + hd] + ad_[d * K_H + hd], 0.2f);
    float alpha = __expf(v - funkey(m[d * K_H + hd])) / den[d * K_H + hd] * scale;
    const float* hp = Hf + (size_t)s * (K_H * Cc) + hd * Cc;
    float* op = concat ? (agg + (size_t)d * (K_H * Cc) + hd * Cc)
                       : (agg + (size_t)d * Cc);
    for (int c = 0; c < Cc; ++c) atomicAdd(&op[c], alpha * hp[c]);
  }
}

__global__ void bias_act(const float* __restrict__ agg, const float* __restrict__ b,
                         float* __restrict__ out, int n, int Cc) {
  int t = blockIdx.x * blockDim.x + threadIdx.x;
  if (t >= n) return;
  float v = agg[t] + b[t % Cc];
  out[t] = lrelu(v, 0.01f);
}

// ---------------- group attention head ----------------
__global__ void compute_vqk(const float* __restrict__ Wq, const float* __restrict__ Wk,
                            const float* __restrict__ Ws, float* __restrict__ Vq,
                            float* __restrict__ Vk) {
  int t = blockIdx.x * blockDim.x + threadIdx.x;   // hd*OC + dd
  if (t >= K_H * K_OC) return;
  int hd = t / K_OC, dd = t % K_OC;
  float s1 = 0.f, s2 = 0.f;
  for (int o = 0; o < K_OC; ++o) {
    s1 += Wq[((size_t)hd * K_OC + dd) * K_OC + o] * Ws[hd * 2 * K_OC + o];
    s2 += Wk[((size_t)hd * K_OC + dd) * K_OC + o] * Ws[hd * 2 * K_OC + K_OC + o];
  }
  Vq[t] = s1; Vk[t] = s2;
}

__global__ void pool_sqk(const float* __restrict__ feat, const int* __restrict__ fidx,
                         const float* __restrict__ Vq, const float* __restrict__ Vk,
                         float* __restrict__ sq, float* __restrict__ sk) {
  int t = blockIdx.x * blockDim.x + threadIdx.x;   // p*H + hd
  if (t >= K_M * K_H) return;
  int p = t / K_H, hd = t % K_H;
  const float* f = feat + (size_t)fidx[p] * 32;
  float s1 = 0.f, s2 = 0.f;
#pragma unroll
  for (int dd = 0; dd < K_OC; ++dd) { s1 += f[dd] * Vq[hd * K_OC + dd]; s2 += f[dd] * Vk[hd * K_OC + dd]; }
  sq[t] = s1; sk[t] = s2;
}

__global__ void weights_kernel(const int* __restrict__ gidx, const int* __restrict__ gsrc,
                               const float* __restrict__ sq, const float* __restrict__ sk,
                               const float* __restrict__ bs, float* __restrict__ wout) {
  int t = blockIdx.x * blockDim.x + threadIdx.x;   // g*H + hd
  if (t >= K_G * K_H) return;
  int g = t / K_H, hd = t % K_H;
  int idx = gidx[g];
  float bsv = bs[hd];
  float sqi = sq[idx * K_H + hd];
  float trg = sigm(sqi + sk[idx * K_H + hd] + bsv);
  float gath[K_L - 1];
#pragma unroll
  for (int j = 0; j < K_L - 1; ++j) {
    int nb = gsrc[g * K_L + 1 + j];
    gath[j] = sigm(sqi + sk[nb * K_H + hd] + bsv);
  }
  float ssum = (gath[0] + gath[1] + gath[2]) * (1.f / 3.f);
  // two-way softmax over {ssum, trg}
  float mx = fmaxf(ssum, trg);
  float e0 = __expf(ssum - mx), e1 = __expf(trg - mx);
  float inv = 1.f / (e0 + e1);
  float tw0 = e0 * inv, tw1 = e1 * inv;
  // softmax over neighbors
  float gm = fmaxf(gath[0], fmaxf(gath[1], gath[2]));
  float ge[K_L - 1]; float gs = 0.f;
#pragma unroll
  for (int j = 0; j < K_L - 1; ++j) { ge[j] = __expf(gath[j] - gm); gs += ge[j]; }
  float* wr = wout + (size_t)g * K_L * K_H;
  wr[0 * K_H + hd] = tw1;
#pragma unroll
  for (int j = 0; j < K_L - 1; ++j)
    wr[(1 + j) * K_H + hd] = (ge[j] / gs) * tw0 * (float)(K_L - 1);
}

// ---------------- host orchestration ----------------
extern "C" void kernel_launch(void* const* d_in, const int* in_sizes, int n_in,
                              void* d_out, int out_size, void* d_ws, size_t ws_size,
                              hipStream_t stream) {
  (void)in_sizes; (void)n_in; (void)out_size; (void)ws_size;

  const float* x       = (const float*)d_in[0];
  const int*   ei      = (const int*)  d_in[1];
  const float* a2s_msg = (const float*)d_in[2];
  const float* a2s_hid = (const float*)d_in[3];
  const float* s2a_msg = (const float*)d_in[4];
  const float* s2a_hid = (const float*)d_in[5];
  const float* prefix  = (const float*)d_in[6];
  const int*   fidx    = (const int*)  d_in[7];
  const int*   gidx    = (const int*)  d_in[8];
  const int*   gsrc    = (const int*)  d_in[9];
  const float* g1Wih = (const float*)d_in[10];
  const float* g1Whh = (const float*)d_in[11];
  const float* g1bih = (const float*)d_in[12];
  const float* g1bhh = (const float*)d_in[13];
  const float* g2Wih = (const float*)d_in[14];
  const float* g2Whh = (const float*)d_in[15];
  const float* g2bih = (const float*)d_in[16];
  const float* g2bhh = (const float*)d_in[17];
  const float* Wmat[4]  = {(const float*)d_in[18], (const float*)d_in[22],
                           (const float*)d_in[26], (const float*)d_in[30]};
  const float* asrc[4]  = {(const float*)d_in[19], (const float*)d_in[23],
                           (const float*)d_in[27], (const float*)d_in[31]};
  const float* adst[4]  = {(const float*)d_in[20], (const float*)d_in[24],
                           (const float*)d_in[28], (const float*)d_in[32]};
  const float* bias[4]  = {(const float*)d_in[21], (const float*)d_in[25],
                           (const float*)d_in[29], (const float*)d_in[33]};
  const float* Wq = (const float*)d_in[34];
  const float* Wk = (const float*)d_in[35];
  const float* Ws = (const float*)d_in[36];
  const float* bs = (const float*)d_in[37];

  float* out = (float*)d_out;
  float* h1  = out + (size_t)K_G * K_L * K_H;    // 16384
  float* h2  = h1 + (size_t)K_A * K_GH;          // +650000

  float* ws    = (float*)d_ws;
  float* feat0 = ws;                               // N*32
  float* feat1 = feat0 + (size_t)K_N * 32;         // N*32
  float* hbuf  = feat1 + (size_t)K_N * 32;         // N*128
  float* asb   = hbuf  + (size_t)K_N * 128;        // N*4
  float* adb   = asb   + (size_t)K_N * 4;          // N*4
  float* mb    = adb   + (size_t)K_N * 4;          // N*4 (uint keys)
  float* denb  = mb    + (size_t)K_N * 4;          // N*4
  float* vq    = denb  + (size_t)K_N * 4;          // 128
  float* vk    = vq + K_H * K_OC;                  // 128
  float* sqb   = vk + K_H * K_OC;                  // 4096
  float* skb   = sqb + K_M * K_H;                  // 4096
  _Float16* whbuf = (_Float16*)(skb + K_M * K_H);  // <= 4096 halves (8KB)

  // GRUs (outputs land directly in d_out and feed feature assembly)
  gru_kernel<<<(K_A + 127) / 128, 128, 0, stream>>>(a2s_msg, a2s_hid, g1Wih, g1Whh, g1bih, g1bhh, h1, K_A);
  gru_kernel<<<(K_B + 127) / 128, 128, 0, stream>>>(s2a_msg, s2a_hid, g2Wih, g2Whh, g2bih, g2bhh, h2, K_B);
  assemble_feat<<<(K_N + 255) / 256, 256, 0, stream>>>(x, prefix, h1, h2, feat0);

  float* fin = feat0;
  float* fout = feat1;
  const int Ks[4] = {16, 32, 32, 32};
  const int Cs[4] = {8, 8, 8, 32};
  for (int l = 0; l < 4; ++l) {
    const int Cc = Cs[l];
    const int Nc = K_H * Cc;
    const int concat = (l < 3) ? 1 : 0;
    // pack weights into B-fragment order (f16), then WMMA transform
    pack_w<<<(Nc * 32 + 255) / 256, 256, 0, stream>>>(Wmat[l], Ks[l], Nc, whbuf);
    wmma_mm<<<dim3(K_N / 16, Nc / 16), 32, 0, stream>>>(fin, whbuf, Nc, hbuf);
    attn_coef<<<(K_N * K_H + 255) / 256, 256, 0, stream>>>(hbuf, Cc, asrc[l], adst[l], asb, adb);
    // zero max-keys + denominators (contiguous) and aggregation target
    fill_kernel<<<(K_N * 8 + 255) / 256, 256, 0, stream>>>(mb, 0.f, K_N * 8);
    fill_kernel<<<(K_N * 32 + 255) / 256, 256, 0, stream>>>(fout, 0.f, K_N * 32);
    edge_max<<<(K_ET + 255) / 256, 256, 0, stream>>>(ei, asb, adb, (unsigned*)mb);
    edge_expsum<<<(K_ET + 255) / 256, 256, 0, stream>>>(ei, asb, adb, (const unsigned*)mb, denb);
    edge_aggr<<<(K_ET + 255) / 256, 256, 0, stream>>>(ei, asb, adb, (const unsigned*)mb, denb,
                                                      hbuf, Cc, fout, concat, concat ? 1.f : 0.25f);
    bias_act<<<(K_N * 32 + 255) / 256, 256, 0, stream>>>(fout, bias[l], fout, K_N * 32, 32);
    float* t = fin; fin = fout; fout = t;
  }

  // final group-attention head (scores computed on the fly, never materialized)
  compute_vqk<<<1, 128, 0, stream>>>(Wq, Wk, Ws, vq, vk);
  pool_sqk<<<(K_M * K_H + 255) / 256, 256, 0, stream>>>(fin, fidx, vq, vk, sqb, skb);
  weights_kernel<<<(K_G * K_H + 255) / 256, 256, 0, stream>>>(gidx, gsrc, sqb, skb, bs, out);
}
